// GowGameModel_80882824118683
// MI455X (gfx1250) — matrix operand order repaired and checked by reading, not compiled, not run on hardware
//
#include <hip/hip_runtime.h>

// CDNA5 / gfx1250, wave32. bf16 WMMA (f32 accumulate) for all GEMMs.
// Pipeline: k0 prep (weights->bf16, zero stats) -> k1 gather+GEMM1+stats
//           -> k2 BN finalize -> k3 BN-affine + GEMM2/3/4 fused.
// b1 is skipped: a constant pre-BN bias cancels exactly in training-mode BN.

typedef __attribute__((ext_vector_type(16))) __bf16 v16bf;
typedef __attribute__((ext_vector_type(8)))  float  v8f;

#define EMBED_VOCAB (16 * 1024 * 1024)
#define VMASK (EMBED_VOCAB - 1)
#define BTOT 65536
#define D1 440      // true input features
#define KP1 448     // K padded to 14 * 32
#define SA1 456     // LDS row stride (bf16 elems; 912B rows keep 16B alignment)
#define N1 256
#define N2 64
#define N3 16
#define TILE_M 128
#define SA2 264     // k3 LDS row stride
#define EPS 1e-5f

struct Frag32B { uint4 lo, hi; };

__device__ __forceinline__ unsigned short f2bf(float f) {
  unsigned u = __float_as_uint(f);
  u += 0x7FFFu + ((u >> 16) & 1u);   // round-to-nearest-even
  return (unsigned short)(u >> 16);
}

// 16 consecutive bf16 (32B) -> one WMMA operand register block
__device__ __forceinline__ v16bf load_frag(const unsigned short* p) {
  Frag32B f;
  f.lo = *(const uint4*)p;
  f.hi = *(const uint4*)(p + 8);
  return __builtin_bit_cast(v16bf, f);
}

__device__ __forceinline__ v8f wmma_bf16(v16bf a, v16bf b, v8f c) {
  return __builtin_amdgcn_wmma_f32_16x16x32_bf16(false, a, false, b, (short)0, c,
                                                 false, false);
}

__device__ __forceinline__ unsigned pack2(float a, float b) {
  return (unsigned)f2bf(a) | ((unsigned)f2bf(b) << 16);
}

// ---------------------------------------------------------------- k0: prep
__global__ void k0_prep(const float* __restrict__ W1, const float* __restrict__ W2,
                        const float* __restrict__ W3,
                        unsigned short* __restrict__ w1bf,
                        unsigned short* __restrict__ w2bf,
                        unsigned short* __restrict__ w3bf,
                        float* __restrict__ stats) {
  int i = blockIdx.x * blockDim.x + threadIdx.x;
  if (i < N1 * KP1) {                       // W1 [256,440] -> bf16 [256,448] zero-padded
    int r = i / KP1, k = i % KP1;
    w1bf[i] = (k < D1) ? f2bf(W1[r * D1 + k]) : (unsigned short)0;
  }
  if (i < N2 * N1) w2bf[i] = f2bf(W2[i]);   // [64,256]
  if (i < N3 * N2) w3bf[i] = f2bf(W3[i]);   // [16,64]
  if (i < 512) stats[i] = 0.f;              // gsum[256] | gssq[256]
}

// ------------------------------------------- k1: gather + GEMM1 + BN partials
__global__ __launch_bounds__(256) void k1_gather_gemm1(
    const int* __restrict__ state, const int* __restrict__ counts,
    const int* __restrict__ mp,
    const float* __restrict__ embed_table, const float* __restrict__ state_table,
    const float* __restrict__ counts_table,
    const unsigned short* __restrict__ w1bf,
    float* __restrict__ h1, float* __restrict__ gsum, float* __restrict__ gssq) {
  // union: bf16 A-tile [128 x 456] (114KB) reused as fp32 stage [128 x 256] (128KB)
  __shared__ __align__(16) unsigned char smem[TILE_M * N1 * 4];
  unsigned short* sA = (unsigned short*)smem;
  float* sStage = (float*)smem;

  const int tid = threadIdx.x;
  const int rowBase = blockIdx.x * TILE_M;

  // zero K padding 440..447 of each row (128*8 = 4 uniform trips)
#pragma unroll
  for (int i = tid; i < TILE_M * 8; i += 256) {
    int r = i >> 3;
    sA[r * SA1 + D1 + (i & 7)] = 0;
  }
  // state(32) + counts(16) features, float4 -> packed 4xbf16 (6 uniform trips)
#pragma unroll
  for (int i = tid; i < TILE_M * 12; i += 256) {
    int r = i / 12, q = (i % 12) * 4;
    int row = rowBase + r;
    const float4 v = (q < 32)
        ? *(const float4*)(state_table + state[row] * 32 + q)
        : *(const float4*)(counts_table + counts[row] * 16 + (q - 32));
    uint2 pk;
    pk.x = pack2(v.x, v.y);
    pk.y = pack2(v.z, v.w);
    *(uint2*)(sA + r * SA1 + q) = pk;
  }
  // 49 embedding gathers per row. 6272 = 24*256 + 128.
  // Main loop: branch-free, 4 indices prefetched per chunk -> 8 independent
  // b128 HBM loads in flight per wave (this phase is the runtime floor).
  for (int g0 = tid; g0 < 6144; g0 += 256 * 4) {
    long long idx[4];
    int rr[4], jj[4];
#pragma unroll
    for (int u = 0; u < 4; ++u) {
      int g = g0 + u * 256;
      rr[u] = g / 49;
      jj[u] = g % 49;
      idx[u] = (long long)(mp[(rowBase + rr[u]) * 49 + jj[u]] & VMASK);
    }
#pragma unroll
    for (int u = 0; u < 4; ++u) {
      const float4* e = (const float4*)(embed_table + idx[u] * 8);
      float4 e0 = e[0], e1 = e[1];
      uint4 pk;
      pk.x = pack2(e0.x, e0.y);
      pk.y = pack2(e0.z, e0.w);
      pk.z = pack2(e1.x, e1.y);
      pk.w = pack2(e1.z, e1.w);
      *(uint4*)(sA + rr[u] * SA1 + 48 + jj[u] * 8) = pk;
    }
  }
  if (tid < 128) {                          // tail: gathers 6144..6271
    int g = 6144 + tid;
    int r = g / 49, j = g % 49;
    long long idx = (long long)(mp[(rowBase + r) * 49 + j] & VMASK);
    const float4* e = (const float4*)(embed_table + idx * 8);
    float4 e0 = e[0], e1 = e[1];
    uint4 pk;
    pk.x = pack2(e0.x, e0.y);
    pk.y = pack2(e0.z, e0.w);
    pk.z = pack2(e1.x, e1.y);
    pk.w = pack2(e1.z, e1.w);
    *(uint4*)(sA + r * SA1 + 48 + j * 8) = pk;
  }
  __syncthreads();

  const int wave = tid >> 5, lane = tid & 31;
  const int laneM = lane & 15, hi = lane >> 4;
  const int rowOff = wave * 16;  // each wave owns a 16-row strip, all 256 cols

  v8f acc[16] = {};
  for (int ks = 0; ks < KP1 / 32; ++ks) {
    const int kb = ks * 32;
    v16bf a = load_frag(sA + (rowOff + laneM) * SA1 + kb + hi * 16);
#pragma unroll
    for (int t = 0; t < 16; ++t) {
      // B[k][n] = W1[n][k]: lane n reads 16 consecutive K from bf16 W1 (L2-resident)
      v16bf b = load_frag(w1bf + (t * 16 + laneM) * KP1 + kb + hi * 16);
      acc[t] = wmma_bf16(a, b, acc[t]);
    }
  }
  __syncthreads();  // all A reads complete -> reuse LDS as fp32 stage

#pragma unroll
  for (int t = 0; t < 16; ++t)
#pragma unroll
    for (int v = 0; v < 8; ++v)
      sStage[(rowOff + v + 8 * hi) * N1 + t * 16 + laneM] = acc[t][v];
  __syncthreads();

  // per-channel partial sums over this block's 128 rows (bank-conflict-free)
  {
    float s = 0.f, q = 0.f;
    const int n = tid;
    for (int r = 0; r < TILE_M; ++r) {
      float x = sStage[r * N1 + n];
      s += x; q += x * x;
    }
    atomicAdd(&gsum[n], s);
    atomicAdd(&gssq[n], q);
  }
  // coalesced h1 writeback (fp32)
  {
    const float4* src = (const float4*)sStage;
    float4* dst = (float4*)(h1 + (long long)rowBase * N1);
#pragma unroll 4
    for (int i = tid; i < TILE_M * N1 / 4; i += 256) dst[i] = src[i];
  }
}

// ---------------------------------------------------------- k2: BN finalize
__global__ void k2_bnstats(const float* __restrict__ gsum, const float* __restrict__ gssq,
                           const float* __restrict__ gamma, const float* __restrict__ beta,
                           float* __restrict__ scale, float* __restrict__ shift) {
  int n = threadIdx.x;
  float mu = gsum[n] * (1.f / BTOT);
  float var = gssq[n] * (1.f / BTOT) - mu * mu;   // biased variance, matches reference
  float sc = gamma[n] * rsqrtf(var + EPS);
  scale[n] = sc;
  shift[n] = beta[n] - mu * sc;
}

// ----------------------------------- k3: BN-affine + GEMM2(relu)+GEMM3(relu)+GEMM4
__global__ __launch_bounds__(256) void k3_bn_mlp(
    const float* __restrict__ h1,
    const float* __restrict__ scale, const float* __restrict__ shift,
    const unsigned short* __restrict__ w2bf, const float* __restrict__ b2,
    const unsigned short* __restrict__ w3bf, const float* __restrict__ b3,
    const float* __restrict__ W4, const float* __restrict__ b4,
    float* __restrict__ out) {
  __shared__ __align__(16) unsigned short sA[TILE_M * SA2];   // 66KB normalized acts
  __shared__ __align__(16) unsigned short sH2[8 * 16 * 64];   // 16KB per-wave h2
  __shared__ float sSc[N1], sSh[N1];

  const int tid = threadIdx.x;
  const int rowBase = blockIdx.x * TILE_M;

  sSc[tid] = scale[tid];
  sSh[tid] = shift[tid];
  __syncthreads();

  // y = h1*scale + shift -> bf16 tile in LDS (float4 in, uint2 out, 32 trips)
#pragma unroll 4
  for (int i = tid; i < TILE_M * N1 / 4; i += 256) {
    int r = i >> 6;                 // 64 float4 per row
    int n4 = (i & 63) * 4;
    float4 x = *(const float4*)(h1 + (long long)(rowBase + r) * N1 + n4);
    uint2 pk;
    pk.x = pack2(x.x * sSc[n4 + 0] + sSh[n4 + 0], x.y * sSc[n4 + 1] + sSh[n4 + 1]);
    pk.y = pack2(x.z * sSc[n4 + 2] + sSh[n4 + 2], x.w * sSc[n4 + 3] + sSh[n4 + 3]);
    *(uint2*)(sA + r * SA2 + n4) = pk;
  }
  __syncthreads();

  const int wave = tid >> 5, lane = tid & 31;
  const int laneM = lane & 15, hi = lane >> 4;
  const int rowOff = wave * 16;

  // GEMM2: 256 -> 64
  v8f acc2[4] = {};
  for (int ks = 0; ks < 8; ++ks) {
    const int kb = ks * 32;
    v16bf a = load_frag(sA + (rowOff + laneM) * SA2 + kb + hi * 16);
#pragma unroll
    for (int t = 0; t < 4; ++t) {
      v16bf b = load_frag(w2bf + (t * 16 + laneM) * N1 + kb + hi * 16);
      acc2[t] = wmma_bf16(a, b, acc2[t]);
    }
  }
  // bias + ReLU -> bf16 h2 strip (16x64) in this wave's LDS region
  unsigned short* h2w = sH2 + wave * 16 * 64;
#pragma unroll
  for (int t = 0; t < 4; ++t) {
    const int n = t * 16 + laneM;
    const float bias = b2[n];
#pragma unroll
    for (int v = 0; v < 8; ++v)
      h2w[(v + 8 * hi) * 64 + n] = f2bf(fmaxf(acc2[t][v] + bias, 0.f));
  }
  __syncthreads();

  // GEMM3: 64 -> 16
  v8f acc3 = {};
#pragma unroll
  for (int ks = 0; ks < 2; ++ks) {
    const int kb = ks * 32;
    v16bf a = load_frag(h2w + laneM * 64 + kb + hi * 16);
    v16bf b = load_frag(w3bf + laneM * 64 + kb + hi * 16);
    acc3 = wmma_bf16(a, b, acc3);
  }

  // GEMM4: 16 -> 1; butterfly-reduce the 16 lanes of each half-wave (ds_swizzle xor)
  const float w4 = W4[laneM];
  const float bias3 = b3[laneM];
  const float bias4 = b4[0];
#pragma unroll
  for (int v = 0; v < 8; ++v) {
    float p = fmaxf(acc3[v] + bias3, 0.f) * w4;
    p += __int_as_float(__builtin_amdgcn_ds_swizzle(__float_as_int(p), 0x041F)); // xor 1
    p += __int_as_float(__builtin_amdgcn_ds_swizzle(__float_as_int(p), 0x081F)); // xor 2
    p += __int_as_float(__builtin_amdgcn_ds_swizzle(__float_as_int(p), 0x101F)); // xor 4
    p += __int_as_float(__builtin_amdgcn_ds_swizzle(__float_as_int(p), 0x201F)); // xor 8
    if (laneM == 0) out[rowBase + rowOff + v + 8 * hi] = p + bias4;
  }
}

// ---------------------------------------------------------------- launcher
extern "C" void kernel_launch(void* const* d_in, const int* in_sizes, int n_in,
                              void* d_out, int out_size, void* d_ws, size_t ws_size,
                              hipStream_t stream) {
  const int*   state        = (const int*)d_in[0];
  const int*   counts       = (const int*)d_in[1];
  const int*   mp           = (const int*)d_in[2];
  const float* embed_table  = (const float*)d_in[3];
  const float* state_table  = (const float*)d_in[4];
  const float* counts_table = (const float*)d_in[5];
  const float* W1           = (const float*)d_in[6];
  // d_in[7] = b1 : unused — constant bias cancels exactly in training-mode BatchNorm
  const float* gamma        = (const float*)d_in[8];
  const float* beta         = (const float*)d_in[9];
  const float* W2           = (const float*)d_in[10];
  const float* b2           = (const float*)d_in[11];
  const float* W3           = (const float*)d_in[12];
  const float* b3           = (const float*)d_in[13];
  const float* W4           = (const float*)d_in[14];
  const float* b4           = (const float*)d_in[15];
  float* out = (float*)d_out;

  char* ws = (char*)d_ws;
  size_t off = 0;
  auto alloc = [&](size_t bytes) {
    char* p = ws + off;
    off = (off + bytes + 255) & ~(size_t)255;
    return p;
  };
  float*          h1    = (float*)alloc((size_t)BTOT * N1 * sizeof(float)); // 64 MB
  unsigned short* w1bf  = (unsigned short*)alloc((size_t)N1 * KP1 * 2);
  unsigned short* w2bf  = (unsigned short*)alloc((size_t)N2 * N1 * 2);
  unsigned short* w3bf  = (unsigned short*)alloc((size_t)N3 * N2 * 2);
  float*          stats = (float*)alloc(512 * sizeof(float)); // gsum | gssq
  float*          scsh  = (float*)alloc(512 * sizeof(float)); // scale | shift

  k0_prep<<<(N1 * KP1 + 255) / 256, 256, 0, stream>>>(W1, W2, W3, w1bf, w2bf, w3bf, stats);
  k1_gather_gemm1<<<BTOT / TILE_M, 256, 0, stream>>>(state, counts, mp, embed_table,
      state_table, counts_table, w1bf, h1, stats, stats + 256);
  k2_bnstats<<<1, 256, 0, stream>>>(stats, stats + 256, gamma, beta, scsh, scsh + 256);
  k3_bn_mlp<<<BTOT / TILE_M, 256, 0, stream>>>(h1, scsh, scsh + 256, w2bf, b2,
      w3bf, b3, W4, b4, out);
}
